// MultiAttentionGATBridge_28913719837513
// MI455X (gfx1250) — compile-verified, matrix operand order checked by reading, not run on hardware
//
#include <hip/hip_runtime.h>
#include <stdint.h>

#define BB 4
#define SS 1024
#define DD 256
#define HH 8
#define LRELU_ALPHA 0.2f
#define LN_EPS 1e-5f

typedef __bf16 v16bf __attribute__((ext_vector_type(16)));
typedef float  v8f   __attribute__((ext_vector_type(8)));

#define WMMA_BF16(a, b, c) \
  __builtin_amdgcn_wmma_f32_16x16x32_bf16(false, (a), false, (b), (short)0, (c), false, false)

// Per-lane K index inside a 32-wide K tile for 16-bit WMMA A/B fragments.
// lane<16: VGPR0..3 = K{0..7} pairs, VGPR4..7 = K{16..23}; lane>=16: +8.
__device__ __forceinline__ int kmap(int e, int hi) {
  int v = e >> 1, p = e & 1;
  int k = (v < 4) ? (2 * v + p) : (16 + 2 * (v - 4) + p);
  return k + (hi << 3);
}

// ---------------------------------------------------------------- packing ---
// X [M x K] f32 row-major -> A fragments [mt][kt][lane][16] bf16
__global__ void pack_a_kernel(const float* __restrict__ X, __bf16* __restrict__ A,
                              int M, int K) {
  int wid  = (blockIdx.x * blockDim.x + threadIdx.x) >> 5;
  int lane = threadIdx.x & 31;
  int KT = K >> 5, MT = M >> 4;
  if (wid >= MT * KT) return;
  int mt = wid / KT, kt = wid % KT;
  int r = lane & 15, hi = lane >> 4;
  const float* row = X + (size_t)(mt * 16 + r) * K + kt * 32;
  __bf16* dst = A + ((size_t)wid * 32 + lane) * 16;
#pragma unroll
  for (int e = 0; e < 16; ++e) dst[e] = (__bf16)row[kmap(e, hi)];
}

// W [K x N] f32 row-major -> B fragments [nt][kt][lane][16] bf16
__global__ void pack_b_kernel(const float* __restrict__ W, __bf16* __restrict__ Bf,
                              int K, int N) {
  int wid  = (blockIdx.x * blockDim.x + threadIdx.x) >> 5;
  int lane = threadIdx.x & 31;
  int KT = K >> 5, NT = N >> 4;
  if (wid >= KT * NT) return;
  int nt = wid / KT, kt = wid % KT;
  int col = lane & 15, hi = lane >> 4;
  __bf16* dst = Bf + ((size_t)wid * 32 + lane) * 16;
#pragma unroll
  for (int e = 0; e < 16; ++e) {
    int k = kt * 32 + kmap(e, hi);
    dst[e] = (__bf16)W[(size_t)k * N + nt * 16 + col];
  }
}

// ------------------------------------------------------------------- GEMM ---
// C[M,N] f32 = A*B (+bias)(+gelu)(+residual). One wave per 16x64 C block:
// each A fragment feeds 4 WMMAs (register blocking over N). N % 64 == 0.
__global__ void gemm_wmma_kernel(const __bf16* __restrict__ A, const __bf16* __restrict__ Bf,
                                 float* __restrict__ C, const float* __restrict__ bias,
                                 const float* __restrict__ residual,
                                 int M, int N, int K, int act) {
  int wid  = (blockIdx.x * blockDim.x + threadIdx.x) >> 5;
  int lane = threadIdx.x & 31;
  int KT = K >> 5, NT4 = N >> 6, MT = M >> 4;
  if (wid >= MT * NT4) return;
  int mt = wid / NT4, n4 = wid % NT4;
  const size_t bstride = (size_t)KT * 512;
  const __bf16* ap = A + (size_t)mt * bstride + (size_t)lane * 16;
  const __bf16* bp0 = Bf + (size_t)(n4 * 4 + 0) * bstride + (size_t)lane * 16;
  const __bf16* bp1 = bp0 + bstride;
  const __bf16* bp2 = bp1 + bstride;
  const __bf16* bp3 = bp2 + bstride;
  v8f acc0 = {}, acc1 = {}, acc2 = {}, acc3 = {};
  for (int kt = 0; kt < KT; ++kt) {
    v16bf a  = *(const v16bf*)ap;
    v16bf b0 = *(const v16bf*)bp0;
    v16bf b1 = *(const v16bf*)bp1;
    v16bf b2 = *(const v16bf*)bp2;
    v16bf b3 = *(const v16bf*)bp3;
    __builtin_prefetch(ap + 512, 0, 1);   // -> global_prefetch_b8
    __builtin_prefetch(bp0 + 512, 0, 1);
    acc0 = WMMA_BF16(a, b0, acc0);
    acc1 = WMMA_BF16(a, b1, acc1);
    acc2 = WMMA_BF16(a, b2, acc2);
    acc3 = WMMA_BF16(a, b3, acc3);
    ap += 512; bp0 += 512; bp1 += 512; bp2 += 512; bp3 += 512;
  }
  v8f accs[4] = {acc0, acc1, acc2, acc3};
  int r = lane & 15, hi = lane >> 4;
#pragma unroll
  for (int j = 0; j < 4; ++j) {
    int col = (n4 * 4 + j) * 16 + r;
    float bval = bias ? bias[col] : 0.f;
#pragma unroll
    for (int idx = 0; idx < 8; ++idx) {
      int row = mt * 16 + idx + hi * 8;
      float v = accs[j][idx] + bval;
      if (act == 1) {  // tanh-approx GELU (jax.nn.gelu default)
        float u = v;
        float t = 0.7978845608028654f * (u + 0.044715f * u * u * u);
        v = 0.5f * u * (1.0f + tanhf(t));
      }
      if (residual) v += residual[(size_t)row * N + col];
      C[(size_t)row * N + col] = v;
    }
  }
}

// ------------------------------------------------------------- GAT pieces ---
// src/dst attention scores: one thread per (b,s,h)
__global__ void scores_kernel(const float* __restrict__ hp, const float* __restrict__ asrc,
                              const float* __restrict__ adst, float* __restrict__ src,
                              float* __restrict__ dst, int dh) {
  int t = blockIdx.x * blockDim.x + threadIdx.x;
  if (t >= BB * SS * HH) return;
  int h = t % HH, s = (t / HH) % SS, b = t / (HH * SS);
  int Ncols = HH * dh;
  const float* row = hp + (size_t)(b * SS + s) * Ncols + h * dh;
  float as = 0.f, ad = 0.f;
  for (int d = 0; d < dh; ++d) {
    float v = row[d];
    as += v * asrc[h * dh + d];
    ad += v * adst[h * dh + d];
  }
  src[((size_t)b * HH + h) * SS + s] = as;
  dst[((size_t)b * HH + h) * SS + s] = ad;
}

// Online softmax row stats (max m / sum z) per (b,h,i). One wave per (b,i),
// all 8 heads in registers, adj row read exactly once, wave32 xor-reduce.
__global__ void rowstats_kernel(const float* __restrict__ adj, const float* __restrict__ src,
                                const float* __restrict__ dst, float* __restrict__ rm,
                                float* __restrict__ rz) {
  int wid  = (blockIdx.x * blockDim.x + threadIdx.x) >> 5;
  int lane = threadIdx.x & 31;
  if (wid >= BB * SS) return;
  int b = wid / SS, i = wid % SS;
  float si[HH], m[HH], z[HH];
#pragma unroll
  for (int h = 0; h < HH; ++h) {
    si[h] = src[((size_t)b * HH + h) * SS + i];
    m[h] = -3.0e38f; z[h] = 0.f;
  }
  const float* arow = adj + ((size_t)b * SS + i) * SS;
  for (int j = lane; j < SS; j += 32) {
    float a = arow[j];
#pragma unroll
    for (int h = 0; h < HH; ++h) {
      float e = si[h] + dst[((size_t)b * HH + h) * SS + j];
      e = (e > 0.f) ? e : LRELU_ALPHA * e;
      if (a <= 0.f) e = -1.0e9f;
      if (e > m[h]) { z[h] = z[h] * __expf(m[h] - e) + 1.f; m[h] = e; }
      else          { z[h] += __expf(e - m[h]); }
    }
  }
#pragma unroll
  for (int off = 16; off > 0; off >>= 1) {
#pragma unroll
    for (int h = 0; h < HH; ++h) {
      float om = __shfl_xor(m[h], off, 32);
      float oz = __shfl_xor(z[h], off, 32);
      float nm = fmaxf(m[h], om);
      z[h] = z[h] * __expf(m[h] - nm) + oz * __expf(om - nm);
      m[h] = nm;
    }
  }
  if (lane == 0) {
#pragma unroll
    for (int h = 0; h < HH; ++h) {
      rm[((size_t)b * HH + h) * SS + i] = m[h];
      rz[((size_t)b * HH + h) * SS + i] = z[h];
    }
  }
}

// Materialize attention probabilities directly as WMMA A-fragments (bf16).
__global__ void pack_p_kernel(const float* __restrict__ adj, const float* __restrict__ src,
                              const float* __restrict__ dst, const float* __restrict__ rm,
                              const float* __restrict__ rz, __bf16* __restrict__ P) {
  const int IT = SS / 16, JT = SS / 32;
  int wid  = (blockIdx.x * blockDim.x + threadIdx.x) >> 5;
  int lane = threadIdx.x & 31;
  if (wid >= BB * HH * IT * JT) return;
  int jt = wid % JT, it = (wid / JT) % IT;
  int h  = (wid / (JT * IT)) % HH, b = wid / (JT * IT * HH);
  int r = lane & 15, hi = lane >> 4;
  int i = it * 16 + r;
  size_t bhs = ((size_t)b * HH + h) * SS;
  float si  = src[bhs + i];
  float mi  = rm[bhs + i];
  float inv = 1.f / rz[bhs + i];
  const float* arow = adj + ((size_t)b * SS + i) * SS + jt * 32;
  const float* drow = dst + bhs + jt * 32;
  __bf16* out = P + ((size_t)wid * 32 + lane) * 16;
#pragma unroll
  for (int e = 0; e < 16; ++e) {
    int k = kmap(e, hi);
    float a  = arow[k];
    float ev = si + drow[k];
    ev = (ev > 0.f) ? ev : LRELU_ALPHA * ev;
    float p = (a > 0.f) ? __expf(ev - mi) * inv : 0.f;
    out[e] = (__bf16)p;
  }
}

// hp [B,S,H,dh] f32 -> per-(b,h) B-fragments over (K=S, N=dh), bf16.
__global__ void pack_hpb_kernel(const float* __restrict__ hp, __bf16* __restrict__ HB, int dh) {
  const int KT = SS / 32;
  int NT = dh >> 4;
  int Ncols = HH * dh;
  int wid  = (blockIdx.x * blockDim.x + threadIdx.x) >> 5;
  int lane = threadIdx.x & 31;
  if (wid >= BB * HH * KT * NT) return;
  int nt = wid % NT, kt = (wid / NT) % KT;
  int h  = (wid / (NT * KT)) % HH, b = wid / (NT * KT * HH);
  int col = lane & 15, hi = lane >> 4;
  __bf16* out = HB + ((size_t)wid * 32 + lane) * 16;
#pragma unroll
  for (int e = 0; e < 16; ++e) {
    int j = kt * 32 + kmap(e, hi);
    out[e] = (__bf16)hp[(size_t)(b * SS + j) * Ncols + h * dh + nt * 16 + col];
  }
}

// Concat layers (dh=32): one wave per (b,h,it). P fragment loaded once,
// feeds both N-tiles of the head (2 WMMAs per A load).
__global__ void attn_gemm_concat_kernel(const __bf16* __restrict__ P,
                                        const __bf16* __restrict__ HB,
                                        float* __restrict__ out) {
  const int IT = SS / 16, JT = SS / 32, NT = 2, dh = 32;
  int wid  = (blockIdx.x * blockDim.x + threadIdx.x) >> 5;
  int lane = threadIdx.x & 31;
  if (wid >= BB * HH * IT) return;
  int it = wid % IT, h = (wid / IT) % HH, b = wid / (IT * HH);
  const __bf16* pp = P  + (((size_t)((b * HH + h) * IT + it) * JT) * 32 + lane) * 16;
  const __bf16* bb = HB + (((size_t)((b * HH + h) * JT) * NT) * 32 + lane) * 16;
  v8f c0 = {}, c1 = {};
  for (int jt = 0; jt < JT; ++jt) {
    v16bf a  = *(const v16bf*)(pp + (size_t)jt * 512);
    v16bf b0 = *(const v16bf*)(bb + (size_t)(jt * NT + 0) * 512);
    v16bf b1 = *(const v16bf*)(bb + (size_t)(jt * NT + 1) * 512);
    __builtin_prefetch(pp + (size_t)(jt + 1) * 512, 0, 1);
    c0 = WMMA_BF16(a, b0, c0);
    c1 = WMMA_BF16(a, b1, c1);
  }
  int r = lane & 15, hi = lane >> 4;
#pragma unroll
  for (int idx = 0; idx < 8; ++idx) {
    int i = it * 16 + idx + hi * 8;
    float* orow = out + (size_t)(b * SS + i) * DD + h * dh;
    orow[r]      = c0[idx];
    orow[16 + r] = c1[idx];
  }
}

// Final layer (dh=256, mean over heads): one wave per (b,it,ngroup).
// 4-way N blocking: each P fragment feeds 4 WMMAs; head loop accumulates
// in registers so the mean is a single scale at the end.
__global__ void attn_gemm_mean_kernel(const __bf16* __restrict__ P,
                                      const __bf16* __restrict__ HB,
                                      float* __restrict__ out) {
  const int IT = SS / 16, JT = SS / 32, NT = 16;  // dh = 256
  int wid  = (blockIdx.x * blockDim.x + threadIdx.x) >> 5;
  int lane = threadIdx.x & 31;
  if (wid >= BB * IT * 4) return;
  int ng = wid % 4, it = (wid / 4) % IT, b = wid / (4 * IT);
  v8f acc[4] = {};
  for (int h = 0; h < HH; ++h) {
    const __bf16* pp = P  + (((size_t)((b * HH + h) * IT + it) * JT) * 32 + lane) * 16;
    const __bf16* bb = HB + (((size_t)((b * HH + h) * JT) * NT + ng * 4) * 32 + lane) * 16;
    for (int jt = 0; jt < JT; ++jt) {
      v16bf a = *(const v16bf*)(pp + (size_t)jt * 512);
      const __bf16* bj = bb + (size_t)jt * NT * 512;
      v16bf b0 = *(const v16bf*)(bj);
      v16bf b1 = *(const v16bf*)(bj + 512);
      v16bf b2 = *(const v16bf*)(bj + 1024);
      v16bf b3 = *(const v16bf*)(bj + 1536);
      __builtin_prefetch(pp + (size_t)(jt + 1) * 512, 0, 1);
      acc[0] = WMMA_BF16(a, b0, acc[0]);
      acc[1] = WMMA_BF16(a, b1, acc[1]);
      acc[2] = WMMA_BF16(a, b2, acc[2]);
      acc[3] = WMMA_BF16(a, b3, acc[3]);
    }
  }
  int r = lane & 15, hi = lane >> 4;
#pragma unroll
  for (int idx = 0; idx < 8; ++idx) {
    int i = it * 16 + idx + hi * 8;
    float* orow = out + (size_t)(b * SS + i) * DD + ng * 64;
#pragma unroll
    for (int j = 0; j < 4; ++j) orow[j * 16 + r] = acc[j][idx] * (1.f / HH);
  }
}

// h = (elu?) (h + att)
__global__ void resid_kernel(float* __restrict__ h, const float* __restrict__ att,
                             int n, int do_elu) {
  int t = blockIdx.x * blockDim.x + threadIdx.x;
  if (t >= n) return;
  float v = h[t] + att[t];
  if (do_elu) v = (v > 0.f) ? v : (__expf(v) - 1.f);
  h[t] = v;
}

// LayerNorm over last dim (256); one wave per row, 8 values per lane.
__global__ void layernorm_kernel(const float* __restrict__ h, const float* __restrict__ g,
                                 const float* __restrict__ bta, float* __restrict__ o) {
  int wid  = (blockIdx.x * blockDim.x + threadIdx.x) >> 5;
  int lane = threadIdx.x & 31;
  if (wid >= BB * SS) return;
  const float* row = h + (size_t)wid * DD;
  float vals[8], s = 0.f, sq = 0.f;
#pragma unroll
  for (int k = 0; k < 8; ++k) {
    float v = row[lane + k * 32];
    vals[k] = v; s += v; sq += v * v;
  }
#pragma unroll
  for (int off = 16; off > 0; off >>= 1) {
    s  += __shfl_xor(s,  off, 32);
    sq += __shfl_xor(sq, off, 32);
  }
  float mu  = s * (1.f / DD);
  float var = sq * (1.f / DD) - mu * mu;
  float rs  = rsqrtf(var + LN_EPS);
  float* orow = o + (size_t)wid * DD;
#pragma unroll
  for (int k = 0; k < 8; ++k) {
    int cidx = lane + k * 32;
    orow[cidx] = (vals[k] - mu) * rs * g[cidx] + bta[cidx];
  }
}

// ------------------------------------------------------------------ driver ---
extern "C" void kernel_launch(void* const* d_in, const int* in_sizes, int n_in,
                              void* d_out, int out_size, void* d_ws, size_t ws_size,
                              hipStream_t stream) {
  (void)in_sizes; (void)n_in; (void)out_size; (void)ws_size;
  const float* adj  = (const float*)d_in[0];
  const float* x    = (const float*)d_in[1];
  const float* W[3]    = {(const float*)d_in[2], (const float*)d_in[5], (const float*)d_in[8]};
  const float* asrc[3] = {(const float*)d_in[3], (const float*)d_in[6], (const float*)d_in[9]};
  const float* adst[3] = {(const float*)d_in[4], (const float*)d_in[7], (const float*)d_in[10]};
  const float* ln_g = (const float*)d_in[11];
  const float* ln_b = (const float*)d_in[12];
  const float* ffw1 = (const float*)d_in[13];
  const float* ffb1 = (const float*)d_in[14];
  const float* ffw2 = (const float*)d_in[15];
  const float* ffb2 = (const float*)d_in[16];

  char* ws = (char*)d_ws;
  size_t off = 0;
  auto alloc = [&](size_t bytes) -> void* {
    void* p = ws + off;
    off += (bytes + 255) & ~(size_t)255;
    return p;
  };
  float*  h    = (float*)alloc((size_t)BB * SS * DD * 4);
  float*  hp   = (float*)alloc((size_t)BB * SS * 2048 * 4);
  float*  hln  = (float*)alloc((size_t)BB * SS * DD * 4);
  float*  att  = (float*)alloc((size_t)BB * SS * DD * 4);
  float*  mid  = (float*)alloc((size_t)BB * SS * 512 * 4);
  float*  srcb = (float*)alloc((size_t)BB * HH * SS * 4);
  float*  dstb = (float*)alloc((size_t)BB * HH * SS * 4);
  float*  rm   = (float*)alloc((size_t)BB * HH * SS * 4);
  float*  rz   = (float*)alloc((size_t)BB * HH * SS * 4);
  __bf16* Af   = (__bf16*)alloc((size_t)4096 * 512 * 2);
  __bf16* Bf   = (__bf16*)alloc((size_t)2048 * 256 * 2);
  __bf16* HB   = (__bf16*)alloc((size_t)BB * HH * SS * 256 * 2);
  __bf16* Pf   = (__bf16*)alloc((size_t)BB * HH * SS * SS * 2);

  hipMemcpyAsync(h, x, (size_t)BB * SS * DD * 4, hipMemcpyDeviceToDevice, stream);

  const int M = BB * SS;
  auto wb = [](int waves) { return (waves + 7) / 8; };  // 8 waves / 256-thr block

  for (int l = 0; l < 3; ++l) {
    int N  = (l < 2) ? DD : HH * DD;
    int dh = N / HH;
    int KT = DD / 32, MT = M / 16, NT = N / 16;
    pack_a_kernel<<<wb(MT * KT), 256, 0, stream>>>(h, Af, M, DD);
    pack_b_kernel<<<wb(KT * NT), 256, 0, stream>>>(W[l], Bf, DD, N);
    gemm_wmma_kernel<<<wb(MT * (N / 64)), 256, 0, stream>>>(Af, Bf, hp, nullptr, nullptr,
                                                            M, N, DD, 0);
    scores_kernel<<<(BB * SS * HH + 255) / 256, 256, 0, stream>>>(hp, asrc[l], adst[l],
                                                                  srcb, dstb, dh);
    rowstats_kernel<<<wb(BB * SS), 256, 0, stream>>>(adj, srcb, dstb, rm, rz);
    pack_p_kernel<<<wb(BB * HH * (SS / 16) * (SS / 32)), 256, 0, stream>>>(adj, srcb, dstb,
                                                                           rm, rz, Pf);
    pack_hpb_kernel<<<wb(BB * HH * (SS / 32) * (dh / 16)), 256, 0, stream>>>(hp, HB, dh);
    if (l < 2) {
      attn_gemm_concat_kernel<<<wb(BB * HH * (SS / 16)), 256, 0, stream>>>(Pf, HB, att);
      resid_kernel<<<(M * DD + 255) / 256, 256, 0, stream>>>(h, att, M * DD, 1);
    } else {
      attn_gemm_mean_kernel<<<wb(BB * (SS / 16) * 4), 256, 0, stream>>>(Pf, HB, att);
      resid_kernel<<<(M * DD + 255) / 256, 256, 0, stream>>>(h, att, M * DD, 0);
    }
  }

  layernorm_kernel<<<wb(BB * SS), 256, 0, stream>>>(h, ln_g, ln_b, hln);

  {  // FFN up: hln @ ffw1 + b1, GELU
    int K = DD, N = 512;
    pack_a_kernel<<<wb((M / 16) * (K / 32)), 256, 0, stream>>>(hln, Af, M, K);
    pack_b_kernel<<<wb((K / 32) * (N / 16)), 256, 0, stream>>>(ffw1, Bf, K, N);
    gemm_wmma_kernel<<<wb((M / 16) * (N / 64)), 256, 0, stream>>>(Af, Bf, mid, ffb1,
                                                                  nullptr, M, N, K, 1);
  }
  {  // FFN down: mid @ ffw2 + b2 + hln -> d_out
    int K = 512, N = DD;
    pack_a_kernel<<<wb((M / 16) * (K / 32)), 256, 0, stream>>>(mid, Af, M, K);
    pack_b_kernel<<<wb((K / 32) * (N / 16)), 256, 0, stream>>>(ffw2, Bf, K, N);
    gemm_wmma_kernel<<<wb((M / 16) * (N / 64)), 256, 0, stream>>>(Af, Bf, (float*)d_out,
                                                                  ffb2, hln, M, N, K, 0);
  }
}
